// LayerNormLSTM_57148834840999
// MI455X (gfx1250) — matrix operand order repaired and checked by reading, not compiled, not run on hardware
//
#include <hip/hip_runtime.h>

typedef unsigned int  u32;
typedef unsigned short u16;
typedef __attribute__((ext_vector_type(16))) __bf16 v16bf;
typedef __attribute__((ext_vector_type(8)))  float  v8f;

constexpr int Bn = 128;   // batch
constexpr int Tn = 256;   // time steps
constexpr int Dn = 768;   // input dim
constexpr int Hn = 768;   // hidden dim
constexpr int G4 = 3072;  // 4*H
constexpr int Kn = 768;   // GEMM K

// ---------- helpers ----------
__device__ __forceinline__ u16 f2bf(float f) {
  union { float f; u32 u; } v; v.f = f;
  u32 u = v.u;
  u += 0x7FFFu + ((u >> 16) & 1u);   // round-to-nearest-even
  return (u16)(u >> 16);
}

// wave32 butterfly + one small LDS stage: 2 barriers total per call
__device__ __forceinline__ float blk_sum(float v, float* red8) {
  #pragma unroll
  for (int m = 16; m >= 1; m >>= 1) v += __shfl_xor(v, m, 32);
  int wave = threadIdx.x >> 5;
  if ((threadIdx.x & 31) == 0) red8[wave] = v;
  __syncthreads();
  float r = 0.f;
  #pragma unroll
  for (int w = 0; w < 8; ++w) r += red8[w];
  __syncthreads();
  return r;
}

// ---------- setup kernels ----------
__global__ void convert_f32_bf16(const float* __restrict__ src, u16* __restrict__ dst, int n) {
  int i = blockIdx.x * blockDim.x + threadIdx.x;
  if (i < n) dst[i] = f2bf(src[i]);
}

__global__ void init_state(u16* __restrict__ hbf, float* __restrict__ c, int n) {
  int i = blockIdx.x * blockDim.x + threadIdx.x;
  if (i < n) { hbf[i] = (u16)0; c[i] = 0.0f; }
}

// ---------- fused dual GEMM: O = A @ W^T, M=128, N=3072, K=768 (bf16 in, f32 acc) ----------
// blockIdx.y selects GEMM 0 or 1. One wave per 16x64 output slab.
// Manual 2x K-unroll with ping-pong A buffers: no register rotation (no v_mov/v_nop
// WAR-hazard churn), and both halves stage all loads before their WMMA burst so the
// scheduler emits partial s_wait_loadcnt and overlaps VMEM with the matrix pipe.
__global__ void __launch_bounds__(256)
gemm2_bf16(const u16* __restrict__ A0, int lda0, const u16* __restrict__ W0, float* __restrict__ O0,
           const u16* __restrict__ A1, int lda1, const u16* __restrict__ W1, float* __restrict__ O1)
{
  const u16* A; const u16* W; float* O; int lda;
  if (blockIdx.y == 0) { A = A0; W = W0; O = O0; lda = lda0; }
  else                 { A = A1; W = W1; O = O1; lda = lda1; }

  int wid   = blockIdx.x * 8 + (threadIdx.x >> 5); // 0..383
  int lane  = threadIdx.x & 31;
  int mtile = wid / 48;                            // 0..7
  int ngrp  = wid % 48;                            // 0..47 -> 4 n-tiles each
  int n0    = ngrp * 64;
  int lo    = lane & 15;
  int hi    = lane >> 4;
  int mrow  = mtile * 16 + lo;
  const u16* arow = A + (size_t)mrow * lda;

  union AF { uint4 q[2]; v16bf v; };
  union BF { uint4 q[2]; v16bf v; };

  v8f acc[4] = {v8f{}, v8f{}, v8f{}, v8f{}};

  // A fragment layout (16-bit A 16x32, ISA 7.12.2): per lane two contiguous
  // 16-byte runs at element offsets (kk + hi*8) and (kk + 16 + hi*8).
  AF a0, a1;
  {
    const uint4* ap = (const uint4*)(arow + hi * 8);
    a0.q[0] = ap[0];
    a0.q[1] = ap[2];
  }

  for (int kk = 0; kk < Kn; kk += 64) {
    // ---- half 0: K block [kk, kk+32) ----
    BF b0[4];
    #pragma unroll
    for (int nt = 0; nt < 4; ++nt) {
      const uint4* wp = (const uint4*)(W + (size_t)(n0 + nt * 16 + lo) * Kn + (kk + hi * 8));
      b0[nt].q[0] = wp[0];
      b0[nt].q[1] = wp[2];
    }
    { // prefetch A for kk+32 (always in range: last kk=704 -> 736 < 768)
      const uint4* ap = (const uint4*)(arow + (kk + 32) + hi * 8);
      a1.q[0] = ap[0];
      a1.q[1] = ap[2];
    }
    #pragma unroll
    for (int nt = 0; nt < 4; ++nt)
      acc[nt] = __builtin_amdgcn_wmma_f32_16x16x32_bf16(
          false, a0.v, false, b0[nt].v, (short)0, acc[nt], false, false);

    // ---- half 1: K block [kk+32, kk+64) ----
    BF b1[4];
    #pragma unroll
    for (int nt = 0; nt < 4; ++nt) {
      const uint4* wp = (const uint4*)(W + (size_t)(n0 + nt * 16 + lo) * Kn + (kk + 32 + hi * 8));
      b1[nt].q[0] = wp[0];
      b1[nt].q[1] = wp[2];
    }
    if (kk + 64 < Kn) { // prefetch A for next iteration
      const uint4* ap = (const uint4*)(arow + (kk + 64) + hi * 8);
      a0.q[0] = ap[0];
      a0.q[1] = ap[2];
    }
    #pragma unroll
    for (int nt = 0; nt < 4; ++nt)
      acc[nt] = __builtin_amdgcn_wmma_f32_16x16x32_bf16(
          false, a1.v, false, b1[nt].v, (short)0, acc[nt], false, false);
  }

  // C/D layout: lane lo = column, VGPR r = row r (+8 for upper lane half)
  int row0 = mtile * 16 + hi * 8;
  #pragma unroll
  for (int nt = 0; nt < 4; ++nt) {
    int col = n0 + nt * 16 + lo;
    #pragma unroll
    for (int r = 0; r < 8; ++r)
      O[(size_t)(row0 + r) * G4 + col] = acc[nt][r];
  }
}

// ---------- fused LayerNorm + gates + cell update (one block per batch row) ----------
__global__ void __launch_bounds__(256)
lngates(const float* __restrict__ rig, const float* __restrict__ rhg,
        const float* __restrict__ gi, const float* __restrict__ bi,
        const float* __restrict__ gh, const float* __restrict__ bh,
        const float* __restrict__ gc, const float* __restrict__ bc,
        const float* __restrict__ cx,
        u16* __restrict__ hy_bf, float* __restrict__ cy_buf,
        float* __restrict__ out_h1, float* __restrict__ out_h2, float* __restrict__ out_c,
        int out_stride)
{
  __shared__ float red8[8];
  int b = blockIdx.x, tid = threadIdx.x;
  const float* rigb = rig + (size_t)b * G4;
  const float* rhgb = rhg + (size_t)b * G4;

  float vi[12], vh[12];
  float s_i = 0.f, q_i = 0.f, s_h = 0.f, q_h = 0.f;
  #pragma unroll
  for (int s = 0; s < 12; ++s) {
    int col = tid + 256 * s;
    vi[s] = rigb[col]; vh[s] = rhgb[col];
    s_i += vi[s]; q_i += vi[s] * vi[s];
    s_h += vh[s]; q_h += vh[s] * vh[s];
  }
  float m_i = blk_sum(s_i, red8) * (1.f / 3072.f);
  float e_i = blk_sum(q_i, red8) * (1.f / 3072.f);
  float m_h = blk_sum(s_h, red8) * (1.f / 3072.f);
  float e_h = blk_sum(q_h, red8) * (1.f / 3072.f);
  float rs_i = rsqrtf(e_i - m_i * m_i + 1e-5f);
  float rs_h = rsqrtf(e_h - m_h * m_h + 1e-5f);

  // cols tid+256*s : gate = s/3, in-gate index hh = tid + 256*(s%3)
  float gate[4][3];
  #pragma unroll
  for (int s = 0; s < 12; ++s) {
    int col = tid + 256 * s;
    float a = (vi[s] - m_i) * rs_i * gi[col] + bi[col];
    float c = (vh[s] - m_h) * rs_h * gh[col] + bh[col];
    gate[s / 3][s % 3] = a + c;
  }

  float craw[3];
  #pragma unroll
  for (int r = 0; r < 3; ++r) {
    int hh = tid + 256 * r;
    float ig = 1.f / (1.f + expf(-gate[0][r]));
    float fg = 1.f / (1.f + expf(-gate[1][r]));
    float cg = tanhf(gate[2][r]);
    craw[r] = fg * cx[(size_t)b * Hn + hh] + ig * cg;
  }
  float sc = 0.f, qc = 0.f;
  #pragma unroll
  for (int r = 0; r < 3; ++r) { sc += craw[r]; qc += craw[r] * craw[r]; }
  float m_c = blk_sum(sc, red8) * (1.f / 768.f);
  float e_c = blk_sum(qc, red8) * (1.f / 768.f);
  float rs_c = rsqrtf(e_c - m_c * m_c + 1e-5f);

  #pragma unroll
  for (int r = 0; r < 3; ++r) {
    int hh = tid + 256 * r;
    float cyv = (craw[r] - m_c) * rs_c * gc[hh] + bc[hh];
    float og  = 1.f / (1.f + expf(-gate[3][r]));
    float hy  = og * tanhf(cyv);
    hy_bf[(size_t)b * Hn + hh]  = f2bf(hy);
    cy_buf[(size_t)b * Hn + hh] = cyv;
    if (out_h1) {
      out_h1[(size_t)b * out_stride + hh] = hy;
      out_h2[(size_t)b * out_stride + hh] = hy;
      out_c [(size_t)b * out_stride + hh] = cyv;
    }
  }
}

// ---------- host ----------
extern "C" void kernel_launch(void* const* d_in, const int* in_sizes, int n_in,
                              void* d_out, int out_size, void* d_ws, size_t ws_size,
                              hipStream_t stream) {
  (void)in_sizes; (void)n_in; (void)out_size; (void)ws_size;
  const float* x    = (const float*)d_in[0];
  const float* w_ih = (const float*)d_in[1];
  const float* w_hh = (const float*)d_in[2];
  const float* g_i  = (const float*)d_in[3];
  const float* b_i  = (const float*)d_in[4];
  const float* g_h  = (const float*)d_in[5];
  const float* b_h  = (const float*)d_in[6];
  const float* g_c  = (const float*)d_in[7];
  const float* b_c  = (const float*)d_in[8];
  float* outp = (float*)d_out;

  char* ws = (char*)d_ws;
  size_t off = 0;
  auto alloc = [&](size_t bytes) -> void* {
    void* p = ws + off;
    off = (off + bytes + 255) & ~(size_t)255;
    return p;
  };

  u16*   wih_bf = (u16*)  alloc((size_t)2 * G4 * Dn * sizeof(u16));
  u16*   whh_bf = (u16*)  alloc((size_t)2 * G4 * Hn * sizeof(u16));
  u16*   x_bf   = (u16*)  alloc((size_t)Bn * Tn * Dn * sizeof(u16));
  float* rig0   = (float*)alloc((size_t)Bn * G4 * sizeof(float));
  float* rhg0   = (float*)alloc((size_t)Bn * G4 * sizeof(float));
  float* rig1   = (float*)alloc((size_t)Bn * G4 * sizeof(float));
  float* rhg1   = (float*)alloc((size_t)Bn * G4 * sizeof(float));
  u16*   hprev  = (u16*)  alloc((size_t)Bn * Hn * sizeof(u16));
  u16*   hy0    = (u16*)  alloc((size_t)Bn * Hn * sizeof(u16));
  float* cprev  = (float*)alloc((size_t)Bn * Hn * sizeof(float));
  float* cy0    = (float*)alloc((size_t)Bn * Hn * sizeof(float));

  // one-time bf16 conversions (weights stay L2-resident: 18.9 MB total)
  {
    int n = 2 * G4 * Dn;
    convert_f32_bf16<<<(n + 255) / 256, 256, 0, stream>>>(w_ih, wih_bf, n);
    convert_f32_bf16<<<(n + 255) / 256, 256, 0, stream>>>(w_hh, whh_bf, n);
    int nx = Bn * Tn * Dn;
    convert_f32_bf16<<<(nx + 255) / 256, 256, 0, stream>>>(x, x_bf, nx);
    int ns = Bn * Hn;
    init_state<<<(ns + 255) / 256, 256, 0, stream>>>(hprev, cprev, ns);
  }

  const int BTH = Bn * Tn * Hn;
  for (int t = 0; t < Tn; ++t) {
    // layer 0: ig0 = x_t @ w_ih0^T, hg0 = h_prev @ w_hh0^T (single fused launch)
    gemm2_bf16<<<dim3(48, 2), 256, 0, stream>>>(
        x_bf + (size_t)t * Dn, Tn * Dn, wih_bf, rig0,
        hprev, Hn, whh_bf, rhg0);
    lngates<<<Bn, 256, 0, stream>>>(
        rig0, rhg0, g_i, b_i, g_h, b_h, g_c, b_c,
        cprev, hy0, cy0, nullptr, nullptr, nullptr, 0);
    // layer 1: ig1 = hy0 @ w_ih1^T, hg1 = hy0 @ w_hh1^T
    gemm2_bf16<<<dim3(48, 2), 256, 0, stream>>>(
        hy0, Hn, wih_bf + (size_t)G4 * Dn, rig1,
        hy0, Hn, whh_bf + (size_t)G4 * Hn, rhg1);
    lngates<<<Bn, 256, 0, stream>>>(
        rig1, rhg1, g_i + G4, b_i + G4, g_h + G4, b_h + G4, g_c + Hn, b_c + Hn,
        cy0, hprev, cprev,
        outp + (size_t)t * Hn,                   // output
        outp + (size_t)BTH + (size_t)t * Hn,     // h_states (== output per reference)
        outp + (size_t)2 * BTH + (size_t)t * Hn, // c_states
        Tn * Hn);
  }
}
